// TensorProductWithMemoryOptimizationWithWeight_23424751632775
// MI455X (gfx1250) — compile-verified, matrix operand order checked by reading, not run on hardware
//
#include <hip/hip_runtime.h>
#include <math.h>
#include <stdint.h>

typedef __attribute__((ext_vector_type(16))) _Float16 v16h;
typedef __attribute__((ext_vector_type(8)))  _Float16 v8h;
typedef __attribute__((ext_vector_type(8)))  float    v8f;
typedef __attribute__((ext_vector_type(4)))  unsigned int v4u;
typedef __attribute__((ext_vector_type(8)))  int v8i;
typedef __attribute__((ext_vector_type(4)))  int v4i;

union FragA { v16h v; v8h h2[2]; _Float16 h[16]; };
union Acc   { v8f  v; float f[8]; };

#define WAVES 8

// blob offsets in halves (pre-swizzled WMMA B-fragments, f16)
#define OFF_W1  0
#define OFF_W2  4096
#define OFF_W3  8192
#define OFF_W4  12288
#define OFF_WM1 16384
#define OFF_WM2 20480
#define OFF_WM3 24576
#define OFF_WL0 40960
#define OFF_WL1 49152
#define BLOB_HALVES 57344
#define BLOB_QW (BLOB_HALVES / 4)   // 14336 8-byte elements

// ---------------------------------------------------------------------------
// Weight conversion: f32 row-major -> f16 WMMA B-fragment blob.
// Fragment (kt,nt): 32 lanes x 16 halves, lane holds column n = nt*16+(lane&15),
// half e holds K = kt*32 + (lane>=16?8:0) + (e<8 ? e : e+8)   [16-bit operand layout]
// ---------------------------------------------------------------------------
__global__ __launch_bounds__(256) void wconv_kernel(
    const float* __restrict__ W1,  const float* __restrict__ W2,
    const float* __restrict__ W3,  const float* __restrict__ W4,
    const float* __restrict__ Wm1, const float* __restrict__ Wm2,
    const float* __restrict__ Wm3, const float* __restrict__ Wl0,
    const float* __restrict__ Wl1, _Float16* __restrict__ blob)
{
  int idx = blockIdx.x * 256 + threadIdx.x;
  if (idx >= BLOB_HALVES) return;
  const float* W; int N; int off;
  if      (idx < OFF_W2)  { W = W1;  N = 64;  off = OFF_W1;  }
  else if (idx < OFF_W3)  { W = W2;  N = 64;  off = OFF_W2;  }
  else if (idx < OFF_W4)  { W = W3;  N = 64;  off = OFF_W3;  }
  else if (idx < OFF_WM1) { W = W4;  N = 64;  off = OFF_W4;  }
  else if (idx < OFF_WM2) { W = Wm1; N = 64;  off = OFF_WM1; }
  else if (idx < OFF_WM3) { W = Wm2; N = 64;  off = OFF_WM2; }
  else if (idx < OFF_WL0) { W = Wm3; N = 256; off = OFF_WM3; }
  else if (idx < OFF_WL1) { W = Wl0; N = 64;  off = OFF_WL0; }
  else                    { W = Wl1; N = 64;  off = OFF_WL1; }
  int local = idx - off;
  int NT   = N >> 4;
  int frag = local >> 9;          // /512 halves per fragment
  int kt   = frag / NT;
  int nt   = frag % NT;
  int r    = local & 511;
  int lane = r >> 4;
  int e    = r & 15;
  int k = kt * 32 + ((lane >= 16) ? 8 : 0) + ((e < 8) ? e : (e + 8));
  int n = nt * 16 + (lane & 15);
  blob[idx] = (_Float16)W[k * N + n];
}

// ---------------------------------------------------------------------------
// device helpers
// ---------------------------------------------------------------------------
__device__ __forceinline__ v8f fzero() {
  v8f c = {0.f, 0.f, 0.f, 0.f, 0.f, 0.f, 0.f, 0.f};
  return c;
}

__device__ __forceinline__ v8f wmma32(v16h a, v16h b, v8f c) {
  return __builtin_amdgcn_wmma_f32_16x16x32_f16(false, a, false, b, (short)0, c,
                                                false, false);
}

__device__ __forceinline__ v16h bfrag(const _Float16* base, int NT, int kt,
                                      int nt, int lane) {
  return *(const v16h*)(base + (((kt * NT + nt) << 9) + (lane << 4)));
}

// A-fragment gather from an LDS row (rowbase points at [M-row][0])
__device__ __forceinline__ v16h afrag_lds(const _Float16* rowbase, int kt,
                                          int hi8) {
  FragA r;
  int b0 = kt * 32 + hi8;
  r.h2[0] = *(const v8h*)(rowbase + b0);
  r.h2[1] = *(const v8h*)(rowbase + b0 + 16);
  return r.v;
}

// Build A fragments for a0 = x0*y0, t0 = sum_i x1_i*y1_i, x0, scalars
__device__ __forceinline__ void buildMainFrags(
    const float* __restrict__ x, const float* __restrict__ scal, size_t row,
    int hi8, float y0, float y1a, float y1b, float y1c, FragA A1[2],
    FragA A2[2], FragA A3[2], FragA AS[2])
{
  const float* xr = x + row * 256;
  const float* sr = scal + row * 64;
#pragma unroll
  for (int kt = 0; kt < 2; ++kt) {
#pragma unroll
    for (int g = 0; g < 2; ++g) {
      int u0 = kt * 32 + hi8 + g * 16;
      float bx0[8], bx1[24], bs[8];
      {
        const float4* p = (const float4*)(xr + u0);
        float4 a = p[0], b = p[1];
        bx0[0]=a.x; bx0[1]=a.y; bx0[2]=a.z; bx0[3]=a.w;
        bx0[4]=b.x; bx0[5]=b.y; bx0[6]=b.z; bx0[7]=b.w;
      }
      {
        const float4* p = (const float4*)(xr + 64 + 3 * u0);
#pragma unroll
        for (int q = 0; q < 6; ++q) {
          float4 t = p[q];
          bx1[4*q]=t.x; bx1[4*q+1]=t.y; bx1[4*q+2]=t.z; bx1[4*q+3]=t.w;
        }
      }
      {
        const float4* p = (const float4*)(sr + u0);
        float4 a = p[0], b = p[1];
        bs[0]=a.x; bs[1]=a.y; bs[2]=a.z; bs[3]=a.w;
        bs[4]=b.x; bs[5]=b.y; bs[6]=b.z; bs[7]=b.w;
      }
#pragma unroll
      for (int j = 0; j < 8; ++j) {
        int e = g * 8 + j;
        float xv = bx0[j];
        float xa = bx1[3*j], xb = bx1[3*j+1], xc = bx1[3*j+2];
        A1[kt].h[e] = (_Float16)(xv * y0);
        A2[kt].h[e] = (_Float16)(xa * y1a + xb * y1b + xc * y1c);
        A3[kt].h[e] = (_Float16)xv;
        AS[kt].h[e] = (_Float16)bs[j];
      }
    }
  }
}

// Build A fragments for b_i = x1[:, :, i] * y0 (reloads x1; L1/L2-hot)
__device__ __forceinline__ void buildA4(const float* __restrict__ x, size_t row,
                                        int hi8, float y0, int i, FragA A4[2])
{
  const float* xr = x + row * 256;
#pragma unroll
  for (int kt = 0; kt < 2; ++kt) {
#pragma unroll
    for (int g = 0; g < 2; ++g) {
      int u0 = kt * 32 + hi8 + g * 16;
      float bx1[24];
      const float4* p = (const float4*)(xr + 64 + 3 * u0);
#pragma unroll
      for (int q = 0; q < 6; ++q) {
        float4 t = p[q];
        bx1[4*q]=t.x; bx1[4*q+1]=t.y; bx1[4*q+2]=t.z; bx1[4*q+3]=t.w;
      }
#pragma unroll
      for (int j = 0; j < 8; ++j)
        A4[kt].h[g * 8 + j] = (_Float16)(bx1[3*j + i] * y0);
    }
  }
}

// ---------------------------------------------------------------------------
// main kernel: one wave == 16 batch rows
// ---------------------------------------------------------------------------
__global__ __launch_bounds__(256) void tp_kernel(
    const float* __restrict__ x, const float* __restrict__ y,
    const float* __restrict__ scal, const _Float16* __restrict__ blob,
    float* __restrict__ out, float silu_cst, int nrows)
{
  __shared__ __align__(16) _Float16 ldsW[BLOB_HALVES];       // 112 KB weights
  __shared__ __align__(16) float    ldsY[WAVES][16][4];      // per-wave y tile
  __shared__ __align__(16) _Float16 ldsT[WAVES][16 * 128];   // per-wave transpose buf

  const int tid   = threadIdx.x;
  const int wave  = tid >> 5;
  const int lane  = tid & 31;
  const int nlane = lane & 15;
  const int hi8   = (lane >= 16) ? 8 : 0;

  const float S1   = 0.125f;          // 1/sqrt(64)
  const float S2   = 0.07216878365f;  // 1/sqrt(192)
  const float INV8 = 0.125f;          // 1/sqrt(64) MLP scale
  const float LIN  = 0.08838834765f;  // 1/sqrt(128)

  // ---- stage weight blob into LDS via the Tensor Data Mover ----
#if __has_builtin(__builtin_amdgcn_tensor_load_to_lds) && __has_builtin(__builtin_amdgcn_s_wait_tensorcnt)
  if (wave == 0) {
    unsigned long long ga = (unsigned long long)(uintptr_t)blob;
    unsigned int ldsa = (unsigned int)(uintptr_t)&ldsW[0];
    // D# group 0: count=1 | lds_addr | global_addr[56:0] | type=2
    v4u g0 = { 1u, ldsa, (unsigned int)(ga & 0xFFFFFFFFull),
               (unsigned int)(((ga >> 32) & 0x01FFFFFFull) | (2u << 30)) };
    // D# group 1: 1-D tensor, data_size=8B, dim0 = tile0 = stride0 = 14336
    v8i g1 = { (int)(3u << 16),                       // data_size=3 (8 bytes)
               (int)((BLOB_QW & 0xFFFF) << 16),       // tensor_dim0[15:0]
               (int)((BLOB_QW >> 16) & 0xFFFF),       // tensor_dim0[31:16]
               (int)((BLOB_QW & 0xFFFF) << 16),       // tile_dim0
               0,                                     // tile_dim1/2 unused
               (int)BLOB_QW,                          // tensor_dim0_stride lo
               0, 0 };
    v4i g2 = { 0, 0, 0, 0 };
    v4i g3 = { 0, 0, 0, 0 };
#if __clang_major__ >= 23
    v8i g4 = { 0, 0, 0, 0, 0, 0, 0, 0 };
    __builtin_amdgcn_tensor_load_to_lds(g0, g1, g2, g3, g4, 0);
#else
    __builtin_amdgcn_tensor_load_to_lds(g0, g1, g2, g3, 0);
#endif
    __builtin_amdgcn_s_wait_tensorcnt(0);
  }
#else
  {
    const uint4* src = (const uint4*)blob;
    uint4* dst = (uint4*)ldsW;
    for (int i = tid; i < BLOB_HALVES / 8; i += 256) dst[i] = src[i];
  }
#endif

  const int tiles = (nrows + 15) >> 4;
  const int tglob = blockIdx.x * WAVES + wave;
  const bool active = (tglob < tiles);
  const int tile = active ? tglob : 0;   // clamp: EXEC stays full for WMMA
  const int r0 = tile * 16;

  if (lane < 16) {
    int rr = r0 + lane; if (rr >= nrows) rr = nrows - 1;
    float4 yv = *(const float4*)(y + (size_t)rr * 4);
    ldsY[wave][lane][0] = yv.x; ldsY[wave][lane][1] = yv.y;
    ldsY[wave][lane][2] = yv.z; ldsY[wave][lane][3] = yv.w;
  }
  __syncthreads();

  // per-A-row y values
  const float y0  = ldsY[wave][nlane][0];
  const float y1a = ldsY[wave][nlane][1];
  const float y1b = ldsY[wave][nlane][2];
  const float y1c = ldsY[wave][nlane][3];
  // per-C-slot y1 values (slot v -> row r0+v+hi8)
  float y1s[3][8];
#pragma unroll
  for (int vv = 0; vv < 8; ++vv) {
    y1s[0][vv] = ldsY[wave][vv + hi8][1];
    y1s[1][vv] = ldsY[wave][vv + hi8][2];
    y1s[2][vv] = ldsY[wave][vv + hi8][3];
  }

  size_t rowA = (size_t)(((r0 + nlane) < nrows) ? (r0 + nlane) : (nrows - 1));

  FragA A1[2], A2[2], A3[2], AS[2];
  buildMainFrags(x, scal, rowA, hi8, y0, y1a, y1b, y1c, A1, A2, A3, AS);

  const _Float16* bW1  = &ldsW[OFF_W1];
  const _Float16* bW2  = &ldsW[OFF_W2];
  const _Float16* bW3  = &ldsW[OFF_W3];
  const _Float16* bW4  = &ldsW[OFF_W4];
  const _Float16* bWm1 = &ldsW[OFF_WM1];
  const _Float16* bWm2 = &ldsW[OFF_WM2];
  const _Float16* bWm3 = &ldsW[OFF_WM3];
  const _Float16* bWl0 = &ldsW[OFF_WL0];
  const _Float16* bWl1 = &ldsW[OFF_WL1];

  _Float16* tbuf = &ldsT[wave][0];
  const _Float16* trow64  = tbuf + nlane * 64;
  const _Float16* trow128 = tbuf + nlane * 128;

  // ---- MLP layer 1: h1 = act(scalars @ Wm1 * inv8) ----
#pragma unroll
  for (int nt = 0; nt < 4; ++nt) {
    v8f c = fzero();
    c = wmma32(AS[0].v, bfrag(bWm1, 4, 0, nt, lane), c);
    c = wmma32(AS[1].v, bfrag(bWm1, 4, 1, nt, lane), c);
    Acc a; a.v = c;
#pragma unroll
    for (int vv = 0; vv < 8; ++vv) {
      float t = a.f[vv] * INV8;
      tbuf[(vv + hi8) * 64 + nt * 16 + nlane] =
          (_Float16)((t / (1.0f + __expf(-t))) * silu_cst);
    }
  }
  FragA h1f[2];
  h1f[0].v = afrag_lds(trow64, 0, hi8);
  h1f[1].v = afrag_lds(trow64, 1, hi8);

  // ---- MLP layer 2 ----
#pragma unroll
  for (int nt = 0; nt < 4; ++nt) {
    v8f c = fzero();
    c = wmma32(h1f[0].v, bfrag(bWm2, 4, 0, nt, lane), c);
    c = wmma32(h1f[1].v, bfrag(bWm2, 4, 1, nt, lane), c);
    Acc a; a.v = c;
#pragma unroll
    for (int vv = 0; vv < 8; ++vv) {
      float t = a.f[vv] * INV8;
      tbuf[(vv + hi8) * 64 + nt * 16 + nlane] =
          (_Float16)((t / (1.0f + __expf(-t))) * silu_cst);
    }
  }
  FragA h2f[2];
  h2f[0].v = afrag_lds(trow64, 0, hi8);
  h2f[1].v = afrag_lds(trow64, 1, hi8);

  // ---- P1 = (x0*y0)@W1 ; P2 = t0@W2 ----
  Acc P1[4], P2[4];
#pragma unroll
  for (int nt = 0; nt < 4; ++nt) {
    v8f c = fzero();
    c = wmma32(A1[0].v, bfrag(bW1, 4, 0, nt, lane), c);
    c = wmma32(A1[1].v, bfrag(bW1, 4, 1, nt, lane), c);
    P1[nt].v = c;
  }
#pragma unroll
  for (int nt = 0; nt < 4; ++nt) {
    v8f c = fzero();
    c = wmma32(A2[0].v, bfrag(bW2, 4, 0, nt, lane), c);
    c = wmma32(A2[1].v, bfrag(bW2, 4, 1, nt, lane), c);
    P2[nt].v = c;
  }

  // ---- sc0 = mid0 * w0 -> LDS (f16, 16x128) ----
#pragma unroll
  for (int nt = 0; nt < 8; ++nt) {
    v8f c = fzero();
    c = wmma32(h2f[0].v, bfrag(bWm3, 16, 0, nt, lane), c);
    c = wmma32(h2f[1].v, bfrag(bWm3, 16, 1, nt, lane), c);
    Acc w; w.v = c;
    const Acc& m = (nt < 4) ? P1[nt] : P2[nt - 4];
    float ms = (nt < 4) ? S1 : S2;
#pragma unroll
    for (int vv = 0; vv < 8; ++vv)
      tbuf[(vv + hi8) * 128 + nt * 16 + nlane] =
          (_Float16)((m.f[vv] * ms) * (w.f[vv] * INV8));
  }

  // ---- o0 = sc0 @ Wl0 * lin_s ----
  {
    FragA sf[4];
#pragma unroll
    for (int kt = 0; kt < 4; ++kt) sf[kt].v = afrag_lds(trow128, kt, hi8);
#pragma unroll
    for (int nt = 0; nt < 4; ++nt) {
      v8f c = fzero();
#pragma unroll
      for (int kt = 0; kt < 4; ++kt)
        c = wmma32(sf[kt].v, bfrag(bWl0, 4, kt, nt, lane), c);
      Acc o; o.v = c;
#pragma unroll
      for (int vv = 0; vv < 8; ++vv) {
        int rr = r0 + vv + hi8;
        if (active && rr < nrows)
          out[(size_t)rr * 256 + nt * 16 + nlane] = o.f[vv] * LIN;
      }
    }
  }

  // ---- P3 = x0@W3 ; gate tiles for w1 ----
  Acc P3[4];
#pragma unroll
  for (int nt = 0; nt < 4; ++nt) {
    v8f c = fzero();
    c = wmma32(A3[0].v, bfrag(bW3, 4, 0, nt, lane), c);
    c = wmma32(A3[1].v, bfrag(bW3, 4, 1, nt, lane), c);
    P3[nt].v = c;
  }
  Acc wA[4], wB[4];
#pragma unroll
  for (int t = 0; t < 4; ++t) {
    v8f c = fzero();
    c = wmma32(h2f[0].v, bfrag(bWm3, 16, 0, 8 + t, lane), c);
    c = wmma32(h2f[1].v, bfrag(bWm3, 16, 1, 8 + t, lane), c);
    wA[t].v = c;
    c = fzero();
    c = wmma32(h2f[0].v, bfrag(bWm3, 16, 0, 12 + t, lane), c);
    c = wmma32(h2f[1].v, bfrag(bWm3, 16, 1, 12 + t, lane), c);
    wB[t].v = c;
  }

  // ---- per i: Q = (x1_i*y0)@W4 ; sc1_i ; o1_i = sc1_i @ Wl1 * lin_s ----
#pragma unroll
  for (int i = 0; i < 3; ++i) {
    FragA A4[2];
    buildA4(x, rowA, hi8, y0, i, A4);
    Acc Q[4];
#pragma unroll
    for (int nt = 0; nt < 4; ++nt) {
      v8f c = fzero();
      c = wmma32(A4[0].v, bfrag(bW4, 4, 0, nt, lane), c);
      c = wmma32(A4[1].v, bfrag(bW4, 4, 1, nt, lane), c);
      Q[nt].v = c;
    }
#pragma unroll
    for (int nt = 0; nt < 4; ++nt) {
#pragma unroll
      for (int vv = 0; vv < 8; ++vv) {
        float m1 = S1 * P3[nt].f[vv] * y1s[i][vv];            // out1a
        tbuf[(vv + hi8) * 128 + nt * 16 + nlane] =
            (_Float16)(m1 * (wA[nt].f[vv] * INV8));
        float m2 = S1 * Q[nt].f[vv];                          // out1b
        tbuf[(vv + hi8) * 128 + (nt + 4) * 16 + nlane] =
            (_Float16)(m2 * (wB[nt].f[vv] * INV8));
      }
    }
    FragA sg[4];
#pragma unroll
    for (int kt = 0; kt < 4; ++kt) sg[kt].v = afrag_lds(trow128, kt, hi8);
#pragma unroll
    for (int nt = 0; nt < 4; ++nt) {
      v8f c = fzero();
#pragma unroll
      for (int kt = 0; kt < 4; ++kt)
        c = wmma32(sg[kt].v, bfrag(bWl1, 4, kt, nt, lane), c);
      Acc o; o.v = c;
#pragma unroll
      for (int vv = 0; vv < 8; ++vv) {
        int rr = r0 + vv + hi8;
        if (active && rr < nrows)
          out[(size_t)rr * 256 + 64 + 3 * (nt * 16 + nlane) + i] =
              o.f[vv] * LIN;
      }
    }
  }
}

// ---------------------------------------------------------------------------
extern "C" void kernel_launch(void* const* d_in, const int* in_sizes, int n_in,
                              void* d_out, int out_size, void* d_ws,
                              size_t ws_size, hipStream_t stream) {
  const float* x   = (const float*)d_in[0];
  const float* y   = (const float*)d_in[1];
  const float* sc  = (const float*)d_in[2];
  const float* W1  = (const float*)d_in[3];
  const float* W2  = (const float*)d_in[4];
  const float* W3  = (const float*)d_in[5];
  const float* W4  = (const float*)d_in[6];
  const float* Wm1 = (const float*)d_in[7];
  const float* Wm2 = (const float*)d_in[8];
  const float* Wm3 = (const float*)d_in[9];
  const float* Wl0 = (const float*)d_in[10];
  const float* Wl1 = (const float*)d_in[11];
  _Float16* blob   = (_Float16*)d_ws;

  int nrows = in_sizes[0] / 256;

  // SILU_CST = 1/sqrt(E[(z*sigmoid(z))^2]) under N(0,1), fine trapezoid quadrature
  double acc = 0.0, nrm = 0.0;
  for (int i = -12000; i <= 12000; ++i) {
    double z = (double)i * 1e-3;
    double w = exp(-0.5 * z * z);
    double s = z / (1.0 + exp(-z));
    acc += w * s * s;
    nrm += w;
  }
  float silu_cst = (float)(1.0 / sqrt(acc / nrm));

  wconv_kernel<<<(BLOB_HALVES + 255) / 256, 256, 0, stream>>>(
      W1, W2, W3, W4, Wm1, Wm2, Wm3, Wl0, Wl1, blob);

  int tiles  = (nrows + 15) / 16;
  int blocks = (tiles + WAVES - 1) / WAVES;
  tp_kernel<<<blocks, 256, 0, stream>>>(x, y, sc, blob, (float*)d_out, silu_cst,
                                        nrows);
}
